// EpisodicMemory_47364899340993
// MI455X (gfx1250) — matrix operand order repaired and checked by reading, not compiled, not run on hardware
//
#include <hip/hip_runtime.h>

// ---------------------------------------------------------------------------
// CDNA5 (gfx1250) implementation: bf16 WMMA GEMM core + fused LSTM scan.
// ---------------------------------------------------------------------------

typedef __bf16 bf16;
typedef __attribute__((ext_vector_type(16))) __bf16 v16bf;
typedef __attribute__((ext_vector_type(8)))  __bf16 v8bf;
typedef __attribute__((ext_vector_type(8)))  float  v8f;
typedef __attribute__((ext_vector_type(8)))  unsigned int v8u;

#define E_   32
#define B_   64
#define D_   896
#define K_   64
#define H_   224
#define G4   896      // 4*H
#define KV_  3072
#define ALPHA_ 5.0e-4f

#define GSL(i, n) for (long long i = (long long)blockIdx.x * blockDim.x + threadIdx.x; \
                       i < (n); i += (long long)gridDim.x * blockDim.x)

__device__ __forceinline__ float sanf(float x, float lo, float hi) {
  if (x != x) return 0.0f;           // NaN -> 0
  x = x > hi ? hi : x;               // +inf -> hi
  x = x < lo ? lo : x;               // -inf -> lo
  return x;
}
__device__ __forceinline__ float sigf(float x) { return 1.0f / (1.0f + __expf(-x)); }

__device__ __forceinline__ v8f zero8() {
  v8f z = {0.f, 0.f, 0.f, 0.f, 0.f, 0.f, 0.f, 0.f};
  return z;
}

// A-operand fragment (16x32 bf16, MxK): lane<16 -> row=lane, K = k0+0..15
//                                        lane>=16 -> row=lane-16, K = k0+16..31
__device__ __forceinline__ v16bf load_fragA(const bf16* base, int lda,
                                            int row0, int k0, int lane) {
  const int r  = row0 + (lane & 15);
  const int kb = k0 + ((lane >> 4) << 4);
  const bf16* q = base + (size_t)r * lda + kb;
  v8bf lo = *(const v8bf*)q;
  v8bf hi = *(const v8bf*)(q + 8);
  v16bf f;
#pragma unroll
  for (int i = 0; i < 8; ++i) { f[i] = lo[i]; f[i + 8] = hi[i]; }
  return f;
}

// B-operand fragment (32x16 bf16, KxN) from N-major (N x K row-major) memory:
// lane<16 -> col=lane, K = k0+0..15 ; lane>=16 -> col=lane-16, K = k0+16..31
__device__ __forceinline__ v16bf load_fragB(const bf16* base, int ldb,
                                            int n0, int k0, int lane) {
  const int n  = n0 + (lane & 15);
  const int kb = k0 + ((lane >> 4) << 4);
  const bf16* q = base + (size_t)n * ldb + kb;
  v8u u = *(const v8u*)q;            // 32B contiguous
  return __builtin_bit_cast(v16bf, u);
}

__device__ __forceinline__ v8f wmma_bf16(v16bf a, v16bf b, v8f c) {
  return __builtin_amdgcn_wmma_f32_16x16x32_bf16(false, a, false, b, (short)0, c,
                                                 false, false);
}

// ---------------------------------------------------------------------------
// Generic batched NT GEMM: C[b] = A[b] (MxK, row-major) * Bt[b]^T (Bt: NxK row-major)
// Optional bias (per column) and optional sanitize-clip epilogue.
// One wave per 16x16 C tile, 4 waves / block, grid.y = batch.
// ---------------------------------------------------------------------------
__global__ __launch_bounds__(128) void wmma_gemm_nt(
    const bf16* __restrict__ A, const bf16* __restrict__ Bt, float* __restrict__ C,
    long long sA, long long sB, long long sC,
    int lda, int ldb, int ldc, int M, int N, int Kd,
    const float* __restrict__ bias, float clip_lo, float clip_hi)
{
  const int lane   = threadIdx.x & 31;
  const int wave   = threadIdx.x >> 5;
  const int tilesN = N >> 4;
  const int tiles  = (M >> 4) * tilesN;
  const int tile   = blockIdx.x * (blockDim.x >> 5) + wave;
  if (tile >= tiles) return;
  const int tm = tile / tilesN, tn = tile - tm * tilesN;
  const bf16* Ab = A + (long long)blockIdx.y * sA;
  const bf16* Bb = Bt + (long long)blockIdx.y * sB;
  float*      Cb = C + (long long)blockIdx.y * sC;

  const int row0 = tm << 4, col0 = tn << 4;
  v8f acc = zero8();
  for (int k = 0; k < Kd; k += 32) {
    v16bf af = load_fragA(Ab, lda, row0, k, lane);
    v16bf bf_ = load_fragB(Bb, ldb, col0, k, lane);
    acc = wmma_bf16(af, bf_, acc);
  }
  const int col = col0 + (lane & 15);
  const int rb  = row0 + ((lane >> 4) << 3);
  const float bv = bias ? bias[col] : 0.0f;
  const bool clip = clip_hi > clip_lo;
#pragma unroll
  for (int r = 0; r < 8; ++r) {
    float v = acc[r] + bv;
    if (clip) v = sanf(v, clip_lo, clip_hi);
    Cb[(size_t)(rb + r) * ldc + col] = v;
  }
}

// ---------------------------------------------------------------------------
// Fused bidirectional LSTM recurrence. grid.x = 2 (direction), 1024 threads.
// h (bf16) lives in LDS; cell state c lives in registers (WMMA C-frag layout).
// Each wave owns up to two (16 batch x 16 hidden) tiles and computes all four
// gate tiles for them, so the nonlinearity is fragment-elementwise.
// ---------------------------------------------------------------------------
__global__ __launch_bounds__(1024) void lstm_scan_kernel(
    const float* __restrict__ xg,    // (2, E, B, 4H)  precomputed z@Wi^T + bi + bh
    const bf16*  __restrict__ Whbf,  // (2, 4H, H) bf16
    bf16*        __restrict__ hcat)  // (E*B, 2H) bf16, column offset dir*H
{
  const int dir = blockIdx.x;
  const float* xgd = xg + (size_t)dir * E_ * B_ * G4;
  const bf16*  Wh  = Whbf + (size_t)dir * G4 * H_;
  __shared__ __align__(16) bf16 hl[B_ * H_];
  const int lane = threadIdx.x & 31;
  const int wave = threadIdx.x >> 5;     // 32 waves
  for (int i = threadIdx.x; i < B_ * H_; i += blockDim.x) hl[i] = (bf16)0.0f;
  __syncthreads();

  v8f cst[2]; v8f hv[2];
  cst[0] = zero8(); cst[1] = zero8();
  hv[0] = zero8(); hv[1] = zero8();
  const int tasks[2] = { wave, wave + 32 };   // 56 tasks = 4 batch-tiles x 14 hid-tiles

  for (int e = 0; e < E_; ++e) {
    const int t = dir ? (E_ - 1 - e) : e;
    const float* xgt = xgd + (size_t)t * B_ * G4;
#pragma unroll
    for (int ti = 0; ti < 2; ++ti) {
      const int task = tasks[ti];
      if (task >= 56) continue;
      const int rm = task / 14, cn = task % 14;
      const int row0 = rm << 4, col0 = cn << 4;
      const int col = col0 + (lane & 15);
      const int rb  = row0 + ((lane >> 4) << 3);
      v8f gi, gf, gg, go;
#pragma unroll
      for (int r = 0; r < 8; ++r) {
        const float* xr = xgt + (size_t)(rb + r) * G4 + col;
        gi[r] = xr[0]; gf[r] = xr[H_]; gg[r] = xr[2 * H_]; go[r] = xr[3 * H_];
      }
      for (int k = 0; k < H_; k += 32) {
        v16bf a = load_fragA(hl, H_, row0, k, lane);
        gi = wmma_bf16(a, load_fragB(Wh, H_, col0,          k, lane), gi);
        gf = wmma_bf16(a, load_fragB(Wh, H_, col0 + H_,     k, lane), gf);
        gg = wmma_bf16(a, load_fragB(Wh, H_, col0 + 2 * H_, k, lane), gg);
        go = wmma_bf16(a, load_fragB(Wh, H_, col0 + 3 * H_, k, lane), go);
      }
#pragma unroll
      for (int r = 0; r < 8; ++r) {
        const float iv = sigf(gi[r]), fv = sigf(gf[r]);
        const float gv = tanhf(gg[r]), ov = sigf(go[r]);
        const float cn2 = fv * cst[ti][r] + iv * gv;
        cst[ti][r] = cn2;
        hv[ti][r]  = ov * tanhf(cn2);
      }
    }
    __syncthreads();                       // all reads of old h done
#pragma unroll
    for (int ti = 0; ti < 2; ++ti) {
      const int task = tasks[ti];
      if (task >= 56) continue;
      const int rm = task / 14, cn = task % 14;
      const int col = (cn << 4) + (lane & 15);
      const int rb  = (rm << 4) + ((lane >> 4) << 3);
#pragma unroll
      for (int r = 0; r < 8; ++r) {
        const bf16 h = (bf16)hv[ti][r];
        hl[(rb + r) * H_ + col] = h;
        hcat[((size_t)t * B_ + (rb + r)) * (2 * H_) + dir * H_ + col] = h;
      }
    }
    __syncthreads();                       // new h visible for next step
  }
}

// ---------------------------------------------------------------------------
// Sherman-Morrison rank-1 memory update scan. One block per batch element.
// U (64x64) lives in LDS; M (64x896) stays resident in L2.
// ---------------------------------------------------------------------------
__global__ __launch_bounds__(256) void sm_scan_kernel(
    const float* __restrict__ w_write,   // (B, E, K)
    const float* __restrict__ z_enc,     // (E, B, D)
    float* __restrict__ Mf)              // (B, K, D), pre-initialized
{
  const int b = blockIdx.x;
  const int tid = threadIdx.x;
  __shared__ float U[K_ * K_];
  __shared__ float wv[K_], Uw[K_];
  __shared__ float delta[D_];
  __shared__ float denom_s;
  for (int i = tid; i < K_ * K_; i += 256)
    U[i] = ((i / K_) == (i % K_)) ? (1.0f + 1e-6f) : 0.0f;
  float* M = Mf + (size_t)b * K_ * D_;
  __syncthreads();
  for (int e = 0; e < E_; ++e) {
    if (tid < K_) wv[tid] = w_write[((size_t)b * E_ + e) * K_ + tid];
    __syncthreads();
    if (tid < K_) {
      float s = 0.f;
      for (int j = 0; j < K_; ++j) s += U[tid * K_ + j] * wv[j];
      Uw[tid] = s;
    }
    __syncthreads();
    if (tid == 0) {
      float s = 0.f;
      for (int k = 0; k < K_; ++k) s += wv[k] * Uw[k];
      denom_s = s + 0.01f;                  // + OBS_NOISE_STD^2
    }
    for (int d = tid; d < D_; d += 256) {
      float s = 0.f;
      for (int k = 0; k < K_; ++k) s += wv[k] * M[(size_t)k * D_ + d];
      delta[d] = z_enc[((size_t)e * B_ + b) * D_ + d] - s;
    }
    __syncthreads();
    const float dn = denom_s;
    for (int i = tid; i < K_ * D_; i += 256) {
      const int k = i / D_, d = i - k * D_;
      M[i] = sanf(M[i] + Uw[k] * delta[d] / dn, -1e6f, 1e6f);
    }
    for (int i = tid; i < K_ * K_; i += 256) {
      const int k = i / K_, j = i - k * K_;
      U[i] = sanf(U[i] - Uw[k] * Uw[j] / dn, -1e6f, 1e6f);
    }
    __syncthreads();
  }
}

// --------------------------- elementwise helpers ---------------------------

__global__ void k_convert_bf16(const float* __restrict__ src, bf16* __restrict__ dst,
                               long long n) {
  GSL(i, n) dst[i] = (bf16)src[i];
}

__global__ void k_bias_sum(const float* a0, const float* b0,
                           const float* a1, const float* b1, float* out) {
  const int i = blockIdx.x * blockDim.x + threadIdx.x;
  if (i < G4) { out[i] = a0[i] + b0[i]; out[G4 + i] = a1[i] + b1[i]; }
}

// zn[b,e,d] = clip(z_enc[e,b,d] + 0.1*eps[b,e,d], +-100) -> bf16
__global__ void k_make_zn(const float* __restrict__ z_enc,
                          const float* __restrict__ eps, bf16* __restrict__ zn) {
  GSL(i, (long long)B_ * E_ * D_) {
    const int d = (int)(i % D_);
    const long long be = i / D_;
    const int e = (int)(be % E_);
    const int b = (int)(be / E_);
    const float v = z_enc[((size_t)e * B_ + b) * D_ + d] + 0.1f * eps[i];
    zn[i] = (bf16)sanf(v, -100.f, 100.f);
  }
}

__global__ void k_mf_init(const float* __restrict__ mm, float* __restrict__ Mf) {
  GSL(i, (long long)B_ * K_ * D_) Mf[i] = mm[i % (K_ * D_)];
}

// A = san(clip(M,+-100)); P = ALPHA*A^T ; emit A_bf (K,D), P fp32/bf16 (D,K), Pt_bf (K,D)
__global__ void k_pinv_init(const float* __restrict__ Msrc, bf16* __restrict__ A_bf,
                            float* __restrict__ P, bf16* __restrict__ P_bf,
                            bf16* __restrict__ Pt_bf, int Bn) {
  GSL(i, (long long)Bn * K_ * D_) {
    const long long b = i / ((long long)K_ * D_);
    const long long rem = i % ((long long)K_ * D_);
    const int k = (int)(rem / D_), d = (int)(rem % D_);
    const float a = sanf(Msrc[i], -100.f, 100.f);
    A_bf[i] = (bf16)a;
    const float p = ALPHA_ * a;
    const size_t pidx = (size_t)b * D_ * K_ + (size_t)d * K_ + k;
    P[pidx] = p; P_bf[pidx] = (bf16)p;
    Pt_bf[i] = (bf16)p;
  }
}

// P = san(2P - T2, +-100); refresh P_bf (D,K) and Pt_bf (K,D)
__global__ void k_pinv_update(float* __restrict__ P, const float* __restrict__ T2,
                              bf16* __restrict__ P_bf, bf16* __restrict__ Pt_bf, int Bn) {
  GSL(i, (long long)Bn * D_ * K_) {
    const long long b = i / ((long long)D_ * K_);
    const long long rem = i % ((long long)D_ * K_);
    const int d = (int)(rem / K_), k = (int)(rem % K_);
    const float p = sanf(2.0f * P[i] - T2[i], -100.f, 100.f);
    P[i] = p; P_bf[i] = (bf16)p;
    Pt_bf[(size_t)b * K_ * D_ + (size_t)k * D_ + d] = (bf16)p;
  }
}

// dst[b][c][r] = (bf16) src[b][r][c]
__global__ void k_transpose_convert(const float* __restrict__ src, bf16* __restrict__ dst,
                                    int Bn, int R, int C) {
  GSL(i, (long long)Bn * R * C) {
    const long long b = i / ((long long)R * C);
    const long long rem = i % ((long long)R * C);
    const int r = (int)(rem / C), c = (int)(rem % C);
    dst[(size_t)b * R * C + (size_t)c * R + r] = (bf16)src[i];
  }
}

// (B,E,D) fp32 -> (E,B,D) bf16
__global__ void k_permute_be(const float* __restrict__ src, bf16* __restrict__ dst) {
  GSL(i, (long long)B_ * E_ * D_) {
    const int d = (int)(i % D_);
    const long long be = i / D_;
    const int e = (int)(be % E_);
    const int b = (int)(be / E_);
    dst[((size_t)e * B_ + b) * D_ + d] = (bf16)src[i];
  }
}

// ------------------------------- host side ---------------------------------

static inline dim3 ewg(long long n) {
  long long b = (n + 255) / 256;
  if (b > 4096) b = 4096;
  return dim3((unsigned)b);
}

static void launch_gemm(hipStream_t s, const bf16* A, long long sA, int lda,
                        const bf16* Bt, long long sB, int ldb,
                        float* C, long long sC, int ldc,
                        int M, int N, int Kd, const float* bias,
                        float clo, float chi, int batch) {
  const int tiles = (M / 16) * (N / 16);
  dim3 grid((tiles + 3) / 4, batch);
  wmma_gemm_nt<<<grid, 128, 0, s>>>(A, Bt, C, sA, sB, sC, lda, ldb, ldc,
                                    M, N, Kd, bias, clo, chi);
}

extern "C" void kernel_launch(void* const* d_in, const int* in_sizes, int n_in,
                              void* d_out, int out_size, void* d_ws, size_t ws_size,
                              hipStream_t stream) {
  (void)in_sizes; (void)n_in; (void)out_size; (void)ws_size;
  const float* z           = (const float*)d_in[0];
  const float* eps_write   = (const float*)d_in[1];
  const float* eps_read    = (const float*)d_in[2];
  const float* memory_mean = (const float*)d_in[3];
  const float* w_ih_f      = (const float*)d_in[4];
  const float* w_hh_f      = (const float*)d_in[5];
  const float* b_ih_f      = (const float*)d_in[6];
  const float* b_hh_f      = (const float*)d_in[7];
  const float* w_ih_b      = (const float*)d_in[8];
  const float* w_hh_b      = (const float*)d_in[9];
  const float* b_ih_b      = (const float*)d_in[10];
  const float* b_hh_b      = (const float*)d_in[11];
  const float* proj_w      = (const float*)d_in[12];
  const float* proj_b      = (const float*)d_in[13];
  const float* WM_w        = (const float*)d_in[14];
  const float* WM_b        = (const float*)d_in[15];
  float* out = (float*)d_out;

  char* base = (char*)d_ws;
  size_t off = 0;
  auto alloc = [&](size_t bytes) -> void* {
    void* p = base + off;
    off = (off + bytes + 255) & ~(size_t)255;
    return p;
  };

  const size_t MB = (size_t)E_ * B_;   // 2048 rows in (e,b) order
  bf16*  z_bf    = (bf16*) alloc(MB * D_ * 2);
  bf16*  wih_bf  = (bf16*) alloc((size_t)2 * G4 * D_ * 2);
  bf16*  whh_bf  = (bf16*) alloc((size_t)2 * G4 * H_ * 2);
  bf16*  proj_bf = (bf16*) alloc((size_t)D_ * 2 * H_ * 2);
  bf16*  wm_bf   = (bf16*) alloc((size_t)KV_ * D_ * 2);
  float* bsum    = (float*)alloc((size_t)2 * G4 * 4);
  float* xg      = (float*)alloc((size_t)2 * MB * G4 * 4);   // reused as T2 later
  bf16*  hcat    = (bf16*) alloc(MB * 2 * H_ * 2);
  float* z_enc   = (float*)alloc(MB * D_ * 4);
  bf16*  zn_w    = (bf16*) alloc((size_t)B_ * E_ * D_ * 2);
  bf16*  zn_r    = (bf16*) alloc((size_t)B_ * E_ * D_ * 2);
  // pinv scratch for the (broadcast) prior memory -- batch 1
  bf16*  A0_bf   = (bf16*) alloc((size_t)K_ * D_ * 2);
  float* P0      = (float*)alloc((size_t)D_ * K_ * 4);
  bf16*  P0_bf   = (bf16*) alloc((size_t)D_ * K_ * 2);
  bf16*  Pt0_bf  = (bf16*) alloc((size_t)K_ * D_ * 2);
  float* S0      = (float*)alloc((size_t)K_ * K_ * 4);
  bf16*  St0_bf  = (bf16*) alloc((size_t)K_ * K_ * 2);
  float* T20     = (float*)alloc((size_t)D_ * K_ * 4);
  float* w_write = (float*)alloc((size_t)B_ * E_ * K_ * 4);
  float* Mf      = (float*)alloc((size_t)B_ * K_ * D_ * 4);
  // batched pinv scratch
  bf16*  A_bf    = (bf16*) alloc((size_t)B_ * K_ * D_ * 2);
  float* P       = (float*)alloc((size_t)B_ * D_ * K_ * 4);
  bf16*  P_bf    = (bf16*) alloc((size_t)B_ * D_ * K_ * 2);
  bf16*  Pt_bf   = (bf16*) alloc((size_t)B_ * K_ * D_ * 2);
  float* S       = (float*)alloc((size_t)B_ * K_ * K_ * 4);
  bf16*  St_bf   = (bf16*) alloc((size_t)B_ * K_ * K_ * 2);
  float* T2      = xg;                 // alias: xg dead after the LSTM
  float* w_read  = (float*)alloc((size_t)B_ * E_ * K_ * 4);
  bf16*  w_read_bf = (bf16*)alloc((size_t)B_ * E_ * K_ * 2);
  bf16*  Mft_bf  = (bf16*) alloc((size_t)B_ * D_ * K_ * 2);
  float* z_read  = (float*)alloc((size_t)B_ * E_ * D_ * 4);
  bf16*  z_read_bf = (bf16*)alloc(MB * D_ * 2);

  // --- bf16 conversions of inputs/weights ---
  k_convert_bf16<<<ewg(MB * D_), 256, 0, stream>>>(z, z_bf, (long long)MB * D_);
  k_convert_bf16<<<ewg((long long)G4 * D_), 256, 0, stream>>>(w_ih_f, wih_bf, (long long)G4 * D_);
  k_convert_bf16<<<ewg((long long)G4 * D_), 256, 0, stream>>>(w_ih_b, wih_bf + (size_t)G4 * D_, (long long)G4 * D_);
  k_convert_bf16<<<ewg((long long)G4 * H_), 256, 0, stream>>>(w_hh_f, whh_bf, (long long)G4 * H_);
  k_convert_bf16<<<ewg((long long)G4 * H_), 256, 0, stream>>>(w_hh_b, whh_bf + (size_t)G4 * H_, (long long)G4 * H_);
  k_convert_bf16<<<ewg((long long)D_ * 2 * H_), 256, 0, stream>>>(proj_w, proj_bf, (long long)D_ * 2 * H_);
  k_convert_bf16<<<ewg((long long)KV_ * D_), 256, 0, stream>>>(WM_w, wm_bf, (long long)KV_ * D_);
  k_bias_sum<<<(G4 + 255) / 256, 256, 0, stream>>>(b_ih_f, b_hh_f, b_ih_b, b_hh_b, bsum);

  // --- xg = z @ Wi^T + (bi+bh) for both directions ---
  for (int dir = 0; dir < 2; ++dir)
    launch_gemm(stream, z_bf, 0, D_, wih_bf + (size_t)dir * G4 * D_, 0, D_,
                xg + (size_t)dir * MB * G4, 0, G4, (int)MB, G4, D_,
                bsum + (size_t)dir * G4, 0.f, 0.f, 1);

  // --- bidirectional LSTM recurrence ---
  lstm_scan_kernel<<<2, 1024, 0, stream>>>(xg, whh_bf, hcat);

  // --- projection: z_enc = [hf;hb] @ Wproj^T + b ---
  launch_gemm(stream, hcat, 0, 2 * H_, proj_bf, 0, 2 * H_,
              z_enc, 0, D_, (int)MB, D_, 2 * H_, proj_b, 0.f, 0.f, 1);

  // --- noisy addressing inputs ---
  k_make_zn<<<ewg((long long)B_ * E_ * D_), 256, 0, stream>>>(z_enc, eps_write, zn_w);
  k_make_zn<<<ewg((long long)B_ * E_ * D_), 256, 0, stream>>>(z_enc, eps_read, zn_r);

  // --- pinv of prior memory (batch-invariant -> batch 1, shared) ---
  k_pinv_init<<<ewg((long long)K_ * D_), 256, 0, stream>>>(memory_mean, A0_bf, P0, P0_bf, Pt0_bf, 1);
  for (int it = 0; it < 3; ++it) {
    // S = A @ P  (K x K)       [ (P A) P == P (A P): 14x fewer FLOPs ]
    launch_gemm(stream, A0_bf, 0, D_, Pt0_bf, 0, D_, S0, 0, K_,
                K_, K_, D_, nullptr, -100.f, 100.f, 1);
    k_transpose_convert<<<ewg((long long)K_ * K_), 256, 0, stream>>>(S0, St0_bf, 1, K_, K_);
    // T2 = P @ S  (D x K)
    launch_gemm(stream, P0_bf, 0, K_, St0_bf, 0, K_, T20, 0, K_,
                D_, K_, K_, nullptr, -100.f, 100.f, 1);
    k_pinv_update<<<ewg((long long)D_ * K_), 256, 0, stream>>>(P0, T20, P0_bf, Pt0_bf, 1);
  }
  // w_write = clip(zn_w @ P0, +-1000)   (shared B operand, strideB = 0)
  launch_gemm(stream, zn_w, (long long)E_ * D_, D_, Pt0_bf, 0, D_,
              w_write, (long long)E_ * K_, K_, E_, K_, D_, nullptr, -1000.f, 1000.f, B_);

  // --- Sherman-Morrison memory write scan ---
  k_mf_init<<<ewg((long long)B_ * K_ * D_), 256, 0, stream>>>(memory_mean, Mf);
  sm_scan_kernel<<<B_, 256, 0, stream>>>(w_write, z_enc, Mf);

  // --- batched pinv of posterior memory ---
  k_pinv_init<<<ewg((long long)B_ * K_ * D_), 256, 0, stream>>>(Mf, A_bf, P, P_bf, Pt_bf, B_);
  for (int it = 0; it < 3; ++it) {
    launch_gemm(stream, A_bf, (long long)K_ * D_, D_, Pt_bf, (long long)K_ * D_, D_,
                S, (long long)K_ * K_, K_, K_, K_, D_, nullptr, -100.f, 100.f, B_);
    k_transpose_convert<<<ewg((long long)B_ * K_ * K_), 256, 0, stream>>>(S, St_bf, B_, K_, K_);
    launch_gemm(stream, P_bf, (long long)D_ * K_, K_, St_bf, (long long)K_ * K_, K_,
                T2, (long long)D_ * K_, K_, D_, K_, K_, nullptr, -100.f, 100.f, B_);
    k_pinv_update<<<ewg((long long)B_ * D_ * K_), 256, 0, stream>>>(P, T2, P_bf, Pt_bf, B_);
  }

  // --- read path ---
  launch_gemm(stream, zn_r, (long long)E_ * D_, D_, Pt_bf, (long long)K_ * D_, D_,
              w_read, (long long)E_ * K_, K_, E_, K_, D_, nullptr, -1000.f, 1000.f, B_);
  k_convert_bf16<<<ewg((long long)B_ * E_ * K_), 256, 0, stream>>>(w_read, w_read_bf,
                                                                   (long long)B_ * E_ * K_);
  k_transpose_convert<<<ewg((long long)B_ * K_ * D_), 256, 0, stream>>>(Mf, Mft_bf, B_, K_, D_);
  // z_read = w_read @ Mf   (B, E, D)
  launch_gemm(stream, w_read_bf, (long long)E_ * K_, K_, Mft_bf, (long long)D_ * K_, K_,
              z_read, (long long)E_ * D_, D_, E_, D_, K_, nullptr, 0.f, 0.f, B_);
  k_permute_be<<<ewg((long long)B_ * E_ * D_), 256, 0, stream>>>(z_read, z_read_bf);

  // --- kv = z_read @ WM_w^T + WM_b   (the big one: 2048 x 3072 x 896) ---
  launch_gemm(stream, z_read_bf, 0, D_, wm_bf, 0, D_,
              out, 0, KV_, (int)MB, KV_, D_, WM_b, 0.f, 0.f, 1);
}
